// Attention_24756191494744
// MI455X (gfx1250) — compile-verified, matrix operand order checked by reading, not run on hardware
//
#include <hip/hip_runtime.h>
#include <hip/hip_bf16.h>

// Problem constants (match reference)
constexpr int SEQ = 2048;   // sequence length
constexpr int BB  = 8;      // batch
constexpr int DIM = 1024;   // input dim
constexpr int ATT = 1024;   // attention dim

typedef __attribute__((ext_vector_type(16))) __bf16          v16bf;
typedef __attribute__((ext_vector_type(16))) unsigned short  v16u;
typedef __attribute__((ext_vector_type(8)))  unsigned short  v8u;
typedef __attribute__((ext_vector_type(8)))  float           v8f;

// fp32 -> bf16, round-to-nearest-even
static __device__ __forceinline__ unsigned short f2bf(float f) {
  unsigned int u = __float_as_uint(f);
  u += 0x7FFFu + ((u >> 16) & 1u);
  return (unsigned short)(u >> 16);
}

static __device__ __forceinline__ v16u concat16(v8u a, v8u b) {
  return __builtin_shufflevector(a, b, 0,1,2,3,4,5,6,7,8,9,10,11,12,13,14,15);
}

static __device__ __forceinline__ v8f wmma_bf16(v16u a, v16u b, v8f c) {
  return __builtin_amdgcn_wmma_f32_16x16x32_bf16(
      /*neg_a=*/false, __builtin_bit_cast(v16bf, a),
      /*neg_b=*/false, __builtin_bit_cast(v16bf, b),
      /*c_mod=*/(short)0, c, /*reuse_a=*/false, /*reuse_b=*/false);
}

// ---------------------------------------------------------------------------
// Kernel 0: elementwise fp32 -> bf16 conversion (memory bound, runs once).
// 8 elements per thread, b128 in / b128 out.
// ---------------------------------------------------------------------------
__global__ __launch_bounds__(256) void cvt_bf16(
    const float* __restrict__ src, unsigned short* __restrict__ dst, int n8)
{
  const int i = blockIdx.x * blockDim.x + threadIdx.x;
  if (i >= n8) return;
  const float4* s = (const float4*)src;
  const float4 a = s[2 * i];
  const float4 b = s[2 * i + 1];
  v8u o;
  o[0] = f2bf(a.x); o[1] = f2bf(a.y); o[2] = f2bf(a.z); o[3] = f2bf(a.w);
  o[4] = f2bf(b.x); o[5] = f2bf(b.y); o[6] = f2bf(b.z); o[7] = f2bf(b.w);
  *(v8u*)(dst + 8 * i) = o;
}

// ---------------------------------------------------------------------------
// Kernel 1: fused QKV projection on bf16 inputs.
//   C = xb * Wb^T + bias   (M=16384, N=3072, K=1024)
//   Q -> (B,S,A) bf16, pre-scaled by 1/sqrt(A); K -> (B,S,A); V -> (B,A,S).
// Block: 256 threads (8 waves, 2 row-groups x 4 col-groups).
// Block tile 64x256; wave tile 32x64 (2 A-frags x 4 B-frags -> 8 WMMA/step).
// ---------------------------------------------------------------------------
__global__ __launch_bounds__(256) void qkv_gemm(
    const unsigned short* __restrict__ xb,   // (16384,1024) bf16, row = s*8+b
    const unsigned short* __restrict__ Wb,   // (3072,1024) bf16 [Wq;Wk;Wv]
    const float* __restrict__ bq, const float* __restrict__ bk,
    const float* __restrict__ bv,
    unsigned short* __restrict__ Qo,
    unsigned short* __restrict__ Ko,
    unsigned short* __restrict__ Vo)
{
  const int lane = threadIdx.x & 31;
  const int wave = threadIdx.x >> 5;
  const int wr   = wave >> 2;                       // row group (0..1)
  const int wc   = wave & 3;                        // col group (0..3)
  const int row0 = blockIdx.x * 64 + wr * 32;       // wave's first row
  const int colg = blockIdx.y * 256 + wc * 64;      // global col (0..3071)
  const int mat  = colg >> 10;                      // 0=Q 1=K 2=V
  const int cm0  = colg & 1023;                     // col within matrix

  const int mrow = lane & 15;
  const int koff = (lane >> 4) << 3;                // 0 or 8

  const unsigned short* x0 = xb + (size_t)(row0 + mrow) * DIM;
  const unsigned short* x1 = x0 + (size_t)16 * DIM;
  const unsigned short* w0 = Wb + (size_t)(colg + mrow) * DIM;

  v8f zero8 = {0.f,0.f,0.f,0.f,0.f,0.f,0.f,0.f};
  v8f acc[2][4] = {{zero8, zero8, zero8, zero8}, {zero8, zero8, zero8, zero8}};

  for (int k = 0; k < DIM; k += 32) {
    const v16u a0 = concat16(*(const v8u*)(x0 + k + koff),
                             *(const v8u*)(x0 + k + 16 + koff));
    const v16u a1 = concat16(*(const v8u*)(x1 + k + koff),
                             *(const v8u*)(x1 + k + 16 + koff));
    #pragma unroll
    for (int nt = 0; nt < 4; ++nt) {
      const unsigned short* wrp = w0 + (size_t)nt * 16 * DIM + k;
      const v16u bfv = concat16(*(const v8u*)(wrp + koff),
                                *(const v8u*)(wrp + 16 + koff));
      acc[0][nt] = wmma_bf16(a0, bfv, acc[0][nt]);
      acc[1][nt] = wmma_bf16(a1, bfv, acc[1][nt]);
    }
  }

  // Epilogue: bias (+ softmax scale folded into Q), convert, store.
  const float* bm = (mat == 0) ? bq : (mat == 1) ? bk : bv;
  const float scale = (mat == 0) ? 0.03125f : 1.0f;   // 1/sqrt(1024)
  const int rhi = (lane >> 4) << 3;                   // lanes 16-31: M = vgpr+8
  #pragma unroll
  for (int mi = 0; mi < 2; ++mi) {
    #pragma unroll
    for (int nt = 0; nt < 4; ++nt) {
      const int cn = cm0 + nt * 16 + (lane & 15);
      const float bias = bm[cn];
      #pragma unroll
      for (int v = 0; v < 8; ++v) {
        const int r = row0 + mi * 16 + v + rhi;       // flat row = s*8 + b
        const int s = r >> 3;
        const int b = r & 7;
        const unsigned short h = f2bf((acc[mi][nt][v] + bias) * scale);
        if (mat == 2) {
          Vo[((size_t)b * ATT + cn) * SEQ + s] = h;           // (B,A,S)
        } else {
          unsigned short* O = (mat == 0) ? Qo : Ko;
          O[((size_t)b * SEQ + s) * ATT + cn] = h;            // (B,S,A)
        }
      }
    }
  }
}

// ---------------------------------------------------------------------------
// Kernel 2: causal flash attention.
// Block = 512 threads = 16 waves, one (batch, 16-query tile) per block.
// Wave w owns A-slice [w*64, w*64+64): partial QK^T over its slice,
// cross-wave reduction via ds_add_f32, online softmax by wave 0, PV WMMA.
// Q/K/V live in L2 (96 MB bf16 < 192 MB), so re-reads never hit HBM.
// ---------------------------------------------------------------------------
__global__ __launch_bounds__(512) void flash_attn(
    const unsigned short* __restrict__ Qw,   // (B,S,A) bf16, pre-scaled
    const unsigned short* __restrict__ Kw,   // (B,S,A) bf16
    const unsigned short* __restrict__ Vt,   // (B,A,S) bf16
    float* __restrict__ out)                 // (S,B,A) fp32
{
  __shared__ float          ls[16][32];      // score tile (fp32 reduction)
  __shared__ unsigned short lp[16][32];      // probs (bf16)
  __shared__ float          lalpha[16];
  __shared__ float          lm[16];
  __shared__ float          ll[16];

  const int lane = threadIdx.x & 31;
  const int wave = threadIdx.x >> 5;
  const int qt   = blockIdx.x;               // query tile
  const int b    = blockIdx.y;
  const int as0  = wave * 64;                // this wave's A-slice

  const int mrow = lane & 15;
  const int koff = (lane >> 4) << 3;         // 0 or 8
  const int rhi  = (lane >> 4) << 3;

  const unsigned short* Qb = Qw + ((size_t)b * SEQ + qt * 16 + mrow) * ATT;
  const unsigned short* Kb = Kw + (size_t)b * SEQ * ATT;
  const unsigned short* Vb = Vt + (size_t)b * ATT * SEQ;

  // Q fragments for this wave's 64-wide slice (two K=32 chunks), kept in regs.
  v16u qa[2];
  #pragma unroll
  for (int c = 0; c < 2; ++c) {
    v8u lo = *(const v8u*)(Qb + as0 + c * 32 + koff);
    v8u hi = *(const v8u*)(Qb + as0 + c * 32 + 16 + koff);
    qa[c] = concat16(lo, hi);
  }

  v8f zero8 = {0.f,0.f,0.f,0.f,0.f,0.f,0.f,0.f};
  v8f o[4] = {zero8, zero8, zero8, zero8};

  if (threadIdx.x < 16) { lm[threadIdx.x] = -__builtin_inff(); ll[threadIdx.x] = 0.f; }
  __syncthreads();

  const int qg_last = qt * 16 + 15;
  for (int kb = 0; kb <= qg_last; kb += 32) {
    // ---- zero score tile (512 floats, one per thread) ----
    ((float*)ls)[threadIdx.x] = 0.f;
    __syncthreads();

    // ---- partial scores for this wave's A-slice: S16x32 += Qslice * Kslice^T
    v8f sf[2] = {zero8, zero8};
    #pragma unroll
    for (int nt = 0; nt < 2; ++nt) {
      #pragma unroll
      for (int c = 0; c < 2; ++c) {
        const unsigned short* kr =
            Kb + (size_t)(kb + nt * 16 + mrow) * ATT + as0 + c * 32;
        v8u lo = *(const v8u*)(kr + koff);
        v8u hi = *(const v8u*)(kr + 16 + koff);
        sf[nt] = wmma_bf16(qa[c], concat16(lo, hi), sf[nt]);
      }
    }
    #pragma unroll
    for (int nt = 0; nt < 2; ++nt)
      #pragma unroll
      for (int v = 0; v < 8; ++v)
        atomicAdd(&ls[v + rhi][nt * 16 + (lane & 15)], sf[nt][v]);
    __syncthreads();

    // ---- online softmax with causal mask (wave 0, one lane per query row)
    if (threadIdx.x < 16) {
      const int m  = threadIdx.x;
      const int qg = qt * 16 + m;
      float mx = lm[m];
      #pragma unroll
      for (int j = 0; j < 32; ++j)
        if (kb + j <= qg) mx = fmaxf(mx, ls[m][j]);
      const float alpha = __expf(lm[m] - mx);
      float sum = 0.f;
      #pragma unroll
      for (int j = 0; j < 32; ++j) {
        const float p = (kb + j <= qg) ? __expf(ls[m][j] - mx) : 0.f;
        sum += p;
        lp[m][j] = f2bf(p);
      }
      lm[m] = mx;
      ll[m] = ll[m] * alpha + sum;
      lalpha[m] = alpha;
    }
    __syncthreads();

    // ---- rescale running O, then O += P * Vslice ----
    float al[8];
    #pragma unroll
    for (int v = 0; v < 8; ++v) al[v] = lalpha[v + rhi];
    #pragma unroll
    for (int nt = 0; nt < 4; ++nt)
      #pragma unroll
      for (int v = 0; v < 8; ++v) o[nt][v] *= al[v];

    v8u plo = *(const v8u*)(&lp[mrow][koff]);
    v8u phi = *(const v8u*)(&lp[mrow][16 + koff]);
    const v16u pA = concat16(plo, phi);

    #pragma unroll
    for (int nt = 0; nt < 4; ++nt) {
      const unsigned short* vr =
          Vb + (size_t)(as0 + nt * 16 + (lane & 15)) * SEQ + kb;
      v8u lo = *(const v8u*)(vr + koff);
      v8u hi = *(const v8u*)(vr + 16 + koff);
      o[nt] = wmma_bf16(pA, concat16(lo, hi), o[nt]);
    }
    __syncthreads();
  }

  // ---- normalize by row sums and store (S,B,A) fp32 ----
  float li[8];
  #pragma unroll
  for (int v = 0; v < 8; ++v) li[v] = 1.0f / ll[v + rhi];
  #pragma unroll
  for (int nt = 0; nt < 4; ++nt) {
    const int cn = as0 + nt * 16 + (lane & 15);
    #pragma unroll
    for (int v = 0; v < 8; ++v) {
      const int qg = qt * 16 + v + rhi;
      out[((size_t)qg * BB + b) * ATT + cn] = o[nt][v] * li[v];
    }
  }
}

extern "C" void kernel_launch(void* const* d_in, const int* in_sizes, int n_in,
                              void* d_out, int out_size, void* d_ws, size_t ws_size,
                              hipStream_t stream) {
  (void)in_sizes; (void)n_in; (void)out_size; (void)ws_size;
  const float* x  = (const float*)d_in[0];
  const float* Wq = (const float*)d_in[1];
  const float* bq = (const float*)d_in[2];
  const float* Wk = (const float*)d_in[3];
  const float* bk = (const float*)d_in[4];
  const float* Wv = (const float*)d_in[5];
  const float* bv = (const float*)d_in[6];
  float* out = (float*)d_out;

  const size_t QKV = (size_t)BB * SEQ * ATT;      // 16,777,216 elements
  const size_t WSZ = (size_t)ATT * DIM;           //  1,048,576 elements
  unsigned short* Qw = (unsigned short*)d_ws;     // (B,S,A)
  unsigned short* Kw = Qw + QKV;                  // (B,S,A)
  unsigned short* Vt = Kw + QKV;                  // (B,A,S)
  unsigned short* xb = Vt + QKV;                  // (S*B, D)
  unsigned short* Wb = xb + QKV;                  // (3072, 1024) [Wq;Wk;Wv]

  // ---- one-time fp32 -> bf16 conversion (memory bound) ----
  {
    const int nx8 = (int)(QKV / 8);
    cvt_bf16<<<dim3((nx8 + 255) / 256), dim3(256), 0, stream>>>(x, xb, nx8);
    const int nw8 = (int)(WSZ / 8);
    cvt_bf16<<<dim3((nw8 + 255) / 256), dim3(256), 0, stream>>>(Wq, Wb, nw8);
    cvt_bf16<<<dim3((nw8 + 255) / 256), dim3(256), 0, stream>>>(Wk, Wb + WSZ, nw8);
    cvt_bf16<<<dim3((nw8 + 255) / 256), dim3(256), 0, stream>>>(Wv, Wb + 2 * WSZ, nw8);
  }

  // ---- QKV projection: M=16384, N=3072, K=1024 ----
  dim3 g1(BB * SEQ / 64, 12);    // 256 x 12 blocks, 256 threads
  qkv_gemm<<<g1, dim3(256), 0, stream>>>(xb, Wb, bq, bk, bv, Qw, Kw, Vt);

  // ---- causal flash attention ----
  dim3 g2(SEQ / 16, BB);         // 128 x 8 blocks, 512 threads
  flash_attn<<<g2, dim3(512), 0, stream>>>(Qw, Kw, Vt, out);
}
